// MMOE_21079699489360
// MI455X (gfx1250) — matrix operand order repaired and checked by reading, not compiled, not run on hardware
//
#include <hip/hip_runtime.h>
#include <stdint.h>

// ---------------- problem constants (match reference) ----------------
#define B_ROWS 65536
#define V_SZ   10000
#define EMB    128
#define D_FEAT 1026          // 8*128 + 2
#define NE     6
#define EH     32
#define EO     16
#define NT     2
#define TH     8

// GEMM tiling: combined weight matrix W_big = [We1 (192 cols) | Wg (12 cols)]
#define NCOL   204           // 6*32 + 2*6
#define NNT    13            // ceil(204/16) N tiles
#define NKT    33            // ceil(1026/32) K tiles (K padded to 1056)
#define NPACK  (NKT*NNT*512) // bf16 elements in packed B

// main kernel shape
#define ROWS_PER_WAVE  32
#define WAVES_PER_BLOCK 2
#define ROWS_PER_BLOCK (ROWS_PER_WAVE*WAVES_PER_BLOCK)
#define FROW   272                    // bytes per row in LDS chunk buffer (128 bf16 + pad)
#define FBUF   (ROWS_PER_WAVE*FROW)   // 8704 B, one chunk buffer per wave
#define FWAVE  (2*FBUF)               // double buffered
#define HSTRIDE 212                   // floats per row in epilogue staging
#define SMEM_BYTES 54272              // max(2*FWAVE=34816, 64*HSTRIDE*4=54272)

typedef __bf16 v16bf __attribute__((ext_vector_type(16)));
typedef float  v8f   __attribute__((ext_vector_type(8)));

static __device__ __forceinline__ unsigned short f2bf(float f) {
  unsigned u = __float_as_uint(f);
  u += 0x7FFFu + ((u >> 16) & 1u);      // round-to-nearest-even
  return (unsigned short)(u >> 16);
}

// ---------------- kernel 1: pack [We1|Wg] into WMMA B-fragment layout ----------------
// B frag (16-bit, 32x16): lane L holds column N=L%16; lane<16 -> K_local {0..7,16..23},
// lane>=16 -> K_local {8..15,24..31} (slots 0..7 then 8..15). Stored fragment-major,
// 1024 B per fragment, lane-contiguous 32 B per lane.
// K is permuted: k'<512 -> k'=k (user embs), 512<=k'<1024 -> k=k'+1 (item embs),
// k'=1024 -> k=512 (x[:,4]), k'=1025 -> k=1025 (x[:,9]), k'>=1026 -> zero pad.
__global__ __launch_bounds__(256) void mmoe_pack(
    const float* __restrict__ We1, const float* __restrict__ Wg,
    unsigned short* __restrict__ wpack)
{
  int o = blockIdx.x * 256 + threadIdx.x;
  if (o >= NPACK) return;
  int s    = o & 15;
  int lane = (o >> 4) & 31;
  int frag = o >> 9;
  int kt = frag / NNT, nt = frag % NNT;
  int kp = kt * 32 + ((lane >> 4) << 3) + (s & 7) + ((s >> 3) << 4);
  int n  = nt * 16 + (lane & 15);
  float val = 0.f;
  if (kp < D_FEAT && n < NCOL) {
    int k;
    if (kp < 512)        k = kp;
    else if (kp < 1024)  k = kp + 1;
    else if (kp == 1024) k = 512;
    else                 k = 1025;
    if (n < NE * EH) {
      int e = n >> 5, h = n & 31;
      val = We1[((size_t)e * D_FEAT + k) * EH + h];
    } else {
      int gidx = n - NE * EH;
      int t = gidx / NE, e = gidx % NE;
      val = Wg[((size_t)t * D_FEAT + k) * NE + e];
    }
  }
  wpack[o] = f2bf(val);
}

// ---------------- main kernel helpers ----------------
// gather one 128-wide embedding segment (chunk<8) or the scalar/pad chunk (chunk==8)
static __device__ __forceinline__ void gather_chunk(
    unsigned char* fb, const float* __restrict__ x, const float* __restrict__ emb,
    int chunk, int rowBase, int lane)
{
  if (chunk < 8) {
    const int c = chunk + (chunk >= 4 ? 1 : 0);   // cate_cols = {0,1,2,3,5,6,7,8}
    const float* ebase = emb + (size_t)chunk * V_SZ * EMB;
    #pragma unroll 4
    for (int r = 0; r < ROWS_PER_WAVE; ++r) {
      const int row = rowBase + r;
      const int idx = (int)x[row * 10 + c];       // wave-uniform -> scalar load
      const float4 f = *(const float4*)(ebase + (size_t)idx * EMB + lane * 4);
      uint2 d;
      d.x = (unsigned)f2bf(f.x) | ((unsigned)f2bf(f.y) << 16);
      d.y = (unsigned)f2bf(f.z) | ((unsigned)f2bf(f.w) << 16);
      *(uint2*)(fb + r * FROW + lane * 8) = d;
    }
  } else {
    // one lane per row: k'=1024 <- x[:,4], k'=1025 <- x[:,9], k' 1026..1055 zero
    const int r = lane;
    const int row = rowBase + r;
    unsigned char* p = fb + r * FROW;
    unsigned sc = (unsigned)f2bf(x[row * 10 + 4]) |
                  ((unsigned)f2bf(x[row * 10 + 9]) << 16);
    *(unsigned*)(p)     = sc;
    *(unsigned*)(p + 4) = 0u;
    uint2 z2; z2.x = 0u; z2.y = 0u;
    *(uint2*)(p + 8) = z2;
    uint4 z4; z4.x = z4.y = z4.z = z4.w = 0u;
    *(uint4*)(p + 16) = z4;
    *(uint4*)(p + 32) = z4;
    *(uint4*)(p + 48) = z4;
  }
}

// A frag (16-bit, 16x32): lane L = row L%16; lanes<16: K {0..7,16..23},
// lanes>=16: K {8..15,24..31} -> two ds_load_b128
static __device__ __forceinline__ v16bf load_a(const unsigned char* rowTile, int ks, int lane) {
  const unsigned char* p = rowTile + (lane & 15) * FROW + ks * 64 + ((lane >> 4) << 4);
  union { uint4 q[2]; v16bf v; } u;
  u.q[0] = *(const uint4*)p;
  u.q[1] = *(const uint4*)(p + 32);
  return u.v;
}

static __device__ __forceinline__ v16bf load_b(const uint4* __restrict__ wp4, int frag, int lane) {
  const uint4* p = wp4 + (size_t)frag * 64 + lane * 2;
  union { uint4 q[2]; v16bf v; } u;
  u.q[0] = p[0];
  u.q[1] = p[1];
  return u.v;
}

// ---------------- kernel 2: fused gather + GEMM + MMoE epilogue ----------------
__global__ __launch_bounds__(64) void mmoe_main(
    const float* __restrict__ x,   const float* __restrict__ emb,
    const float* __restrict__ be1, const float* __restrict__ We2,
    const float* __restrict__ be2, const float* __restrict__ Wt1,
    const float* __restrict__ bt1, const float* __restrict__ Wt2,
    const float* __restrict__ bt2, const unsigned short* __restrict__ wpack,
    float* __restrict__ out)
{
  __shared__ __align__(16) unsigned char smem[SMEM_BYTES];
  const int tid  = threadIdx.x;
  const int wave = tid >> 5;
  const int lane = tid & 31;
  const int rowBase = blockIdx.x * ROWS_PER_BLOCK + wave * ROWS_PER_WAVE;
  unsigned char* fwave = smem + wave * FWAVE;
  const uint4* wp4 = (const uint4*)wpack;

  const v8f vzero = {0.f, 0.f, 0.f, 0.f, 0.f, 0.f, 0.f, 0.f};
  v8f acc[2][NNT];
  #pragma unroll
  for (int m = 0; m < 2; ++m)
    #pragma unroll
    for (int nt = 0; nt < NNT; ++nt) acc[m][nt] = vzero;

  gather_chunk(fwave, x, emb, 0, rowBase, lane);   // chunk 0 -> buffer 0

  for (int c = 0; c <= 8; ++c) {
    __syncthreads();                               // gather(c) visible
    if (c < 8)
      gather_chunk(fwave + (((c + 1) & 1) ? FBUF : 0), x, emb, c + 1, rowBase, lane);
    const unsigned char* cur = fwave + ((c & 1) ? FBUF : 0);
    const int nks = (c < 8) ? 4 : 1;               // chunk 8 = scalars, 1 k-step
    for (int ks = 0; ks < nks; ++ks) {
      v16bf a0 = load_a(cur, ks, lane);
      v16bf a1 = load_a(cur + 16 * FROW, ks, lane);
      const int ktAbs = c * 4 + ks;
      #pragma unroll
      for (int nt = 0; nt < NNT; ++nt) {
        v16bf b = load_b(wp4, ktAbs * NNT + nt, lane);
        acc[0][nt] = __builtin_amdgcn_wmma_f32_16x16x32_bf16(
            false, a0, false, b, (short)0, acc[0][nt], false, false);
        acc[1][nt] = __builtin_amdgcn_wmma_f32_16x16x32_bf16(
            false, a1, false, b, (short)0, acc[1][nt], false, false);
      }
    }
  }

  // ---- dump accumulators to LDS (C layout: lane holds M = v + 8*(lane/16), N = lane%16) ----
  __syncthreads();                                 // done reading chunk buffers
  float* hb = (float*)smem;
  #pragma unroll
  for (int mt = 0; mt < 2; ++mt)
    #pragma unroll
    for (int nt = 0; nt < NNT; ++nt)
      #pragma unroll
      for (int v = 0; v < 8; ++v) {
        const int R = wave * ROWS_PER_WAVE + mt * 16 + ((lane >> 4) << 3) + v;
        hb[R * HSTRIDE + nt * 16 + (lane & 15)] = acc[mt][nt][v];
      }
  __syncthreads();

  // ---- per-row epilogue: one thread per row ----
  const int R   = tid;
  const int row = blockIdx.x * ROWS_PER_BLOCK + R;
  const float* hrow = hb + R * HSTRIDE;

  // gate softmax (cols 192..203: t=0 -> 192..197, t=1 -> 198..203)
  float g[2][NE];
  #pragma unroll
  for (int t = 0; t < NT; ++t) {
    float m = -3.4e38f;
    #pragma unroll
    for (int e = 0; e < NE; ++e) m = fmaxf(m, hrow[192 + t * NE + e]);
    float s = 0.f;
    #pragma unroll
    for (int e = 0; e < NE; ++e) {
      float v = __expf(hrow[192 + t * NE + e] - m);
      g[t][e] = v; s += v;
    }
    const float inv = 1.f / s;
    #pragma unroll
    for (int e = 0; e < NE; ++e) g[t][e] *= inv;
  }

  float ti[NT][EO];
  #pragma unroll
  for (int o = 0; o < EO; ++o) { ti[0][o] = 0.f; ti[1][o] = 0.f; }

  for (int e = 0; e < NE; ++e) {
    float hr[EH];
    #pragma unroll
    for (int i = 0; i < EH; ++i)
      hr[i] = fmaxf(hrow[e * EH + i] + be1[e * EH + i], 0.f);
    #pragma unroll
    for (int o = 0; o < EO; ++o) {
      float s = be2[e * EO + o];
      #pragma unroll
      for (int i = 0; i < EH; ++i)
        s = fmaf(hr[i], We2[(e * EH + i) * EO + o], s);   // uniform -> scalar loads
      ti[0][o] = fmaf(g[0][e], s, ti[0][o]);
      ti[1][o] = fmaf(g[1][e], s, ti[1][o]);
    }
  }

  #pragma unroll
  for (int t = 0; t < NT; ++t) {
    float z = bt2[t];
    #pragma unroll
    for (int hh = 0; hh < TH; ++hh) {
      float s = bt1[t * TH + hh];
      #pragma unroll
      for (int o = 0; o < EO; ++o)
        s = fmaf(ti[t][o], Wt1[(t * EO + o) * TH + hh], s);
      z = fmaf(fmaxf(s, 0.f), Wt2[t * TH + hh], z);
    }
    out[t * B_ROWS + row] = 1.f / (1.f + __expf(-z));
  }
}

// ---------------- host entry ----------------
extern "C" void kernel_launch(void* const* d_in, const int* in_sizes, int n_in,
                              void* d_out, int out_size, void* d_ws, size_t ws_size,
                              hipStream_t stream) {
  (void)in_sizes; (void)n_in; (void)out_size; (void)ws_size;
  const float* x   = (const float*)d_in[0];
  const float* emb = (const float*)d_in[1];
  const float* We1 = (const float*)d_in[2];
  const float* be1 = (const float*)d_in[3];
  const float* We2 = (const float*)d_in[4];
  const float* be2 = (const float*)d_in[5];
  const float* Wg  = (const float*)d_in[6];
  const float* Wt1 = (const float*)d_in[7];
  const float* bt1 = (const float*)d_in[8];
  const float* Wt2 = (const float*)d_in[9];
  const float* bt2 = (const float*)d_in[10];
  float* out = (float*)d_out;
  unsigned short* wpack = (unsigned short*)d_ws;   // needs 439,296 B of scratch

  mmoe_pack<<<(NPACK + 255) / 256, 256, 0, stream>>>(We1, Wg, wpack);
  mmoe_main<<<B_ROWS / ROWS_PER_BLOCK, WAVES_PER_BLOCK * 32, 0, stream>>>(
      x, emb, be1, We2, be2, Wt1, bt1, Wt2, bt2, wpack, out);
}
